// BertSelfAttention_12738873000441
// MI455X (gfx1250) — matrix operand order repaired and checked by reading, not compiled, not run on hardware
//
#include <hip/hip_runtime.h>

typedef __bf16 bf16_t;
typedef __attribute__((ext_vector_type(16))) __bf16 v16bf;
typedef __attribute__((ext_vector_type(8)))  __bf16 v8bf;
typedef __attribute__((ext_vector_type(8)))  float  v8f;
typedef __attribute__((ext_vector_type(4)))  unsigned int v4u;

#define B_DIM 4
#define L_DIM 2048
#define D_DIM 1024
// scale = 1/sqrt(HEAD_SIZE=64)
#define SCORE_SCALE 0.125f

union pack2 { unsigned int u; bf16_t h[2]; };
union pack8 { v4u u4; bf16_t h[8]; };

// ---------------------------------------------------------------------------
// Projection GEMM: out = X[M,K] @ W[K,N] + bias,  M=8192, N=K=1024, bf16 out.
// MODE 0: row-major out (B*L, D). Computed as (W_tile^T x X_tile) so each
//         lane's 8 accum elements walk n contiguously -> one b128 store.
// MODE 1: transposed out (B, D, L) for V. Normal orientation; lane's 8 accum
//         elements walk l contiguously -> one b128 store.
// Block: 256 threads (8 waves), tile 64(M) x 128(N), K-step 32.
// ---------------------------------------------------------------------------
template <int MODE>
__global__ __launch_bounds__(256) void qkv_gemm_kernel(
    const float* __restrict__ x, const float* __restrict__ W,
    const float* __restrict__ bias, bf16_t* __restrict__ out)
{
    __shared__ bf16_t a_lds[64][32];    // X tile (M x K) bf16
    __shared__ bf16_t wt_lds[128][32];  // W tile transposed (N x K) bf16

    const int tid  = threadIdx.x;
    const int lane = tid & 31;
    const int wave = tid >> 5;
    const int half = lane >> 4;     // 0 or 1
    const int lrow = lane & 15;

    const int n0 = blockIdx.x * 128;
    const int m0 = blockIdx.y * 64;
    const int m_base = (wave >> 2) * 32;  // 2 wave-rows of 32 (m)
    const int n_base = (wave & 3) * 32;   // 4 wave-cols of 32 (n)

    v8f acc[2][2] = {};

    for (int k0 = 0; k0 < D_DIM; k0 += 32) {
        // stage X tile: 64x32 -> 1024 packed u32, 4 per thread (coalesced in K)
        for (int i = 0; i < 4; ++i) {
            int idx = tid + i * 256;          // u32 index
            int m = idx >> 4, kk2 = idx & 15; // 16 u32 per row
            const float* src = &x[(size_t)(m0 + m) * D_DIM + k0 + kk2 * 2];
            pack2 p; p.h[0] = (bf16_t)src[0]; p.h[1] = (bf16_t)src[1];
            *(unsigned int*)&a_lds[m][kk2 * 2] = p.u;
        }
        // stage W tile transposed: 128x32 -> 2048 packed u32, 8 per thread.
        // pack along kk (two W rows), coalesced along n.
        for (int i = 0; i < 8; ++i) {
            int idx = tid + i * 256;
            int n = idx & 127, kk = (idx >> 7) * 2;
            pack2 p;
            p.h[0] = (bf16_t)W[(size_t)(k0 + kk)     * D_DIM + n0 + n];
            p.h[1] = (bf16_t)W[(size_t)(k0 + kk + 1) * D_DIM + n0 + n];
            *(unsigned int*)&wt_lds[n][kk] = p.u;
        }
        __syncthreads();

        if (MODE == 0) {
            // A = W^T rows (n), B = X columns (m)
            v16bf afrag[2], bfrag[2];
            for (int ni = 0; ni < 2; ++ni) {
                int n = n_base + ni * 16 + lrow;
                afrag[ni] = *(const v16bf*)&wt_lds[n][half * 16];
            }
            for (int mi = 0; mi < 2; ++mi) {
                int m = m_base + mi * 16 + lrow;
                bfrag[mi] = *(const v16bf*)&a_lds[m][half * 16];
            }
            for (int ni = 0; ni < 2; ++ni)
                for (int mi = 0; mi < 2; ++mi)
                    acc[ni][mi] = __builtin_amdgcn_wmma_f32_16x16x32_bf16(
                        false, afrag[ni], false, bfrag[mi],
                        (short)0, acc[ni][mi], false, false);
        } else {
            // A = X rows (m), B = W^T columns (n)
            v16bf afrag[2], bfrag[2];
            for (int mi = 0; mi < 2; ++mi) {
                int m = m_base + mi * 16 + lrow;
                afrag[mi] = *(const v16bf*)&a_lds[m][half * 16];
            }
            for (int ni = 0; ni < 2; ++ni) {
                int n = n_base + ni * 16 + lrow;
                bfrag[ni] = *(const v16bf*)&wt_lds[n][half * 16];
            }
            for (int mi = 0; mi < 2; ++mi)
                for (int ni = 0; ni < 2; ++ni)
                    acc[mi][ni] = __builtin_amdgcn_wmma_f32_16x16x32_bf16(
                        false, afrag[mi], false, bfrag[ni],
                        (short)0, acc[mi][ni], false, false);
        }
        __syncthreads();
    }

    if (MODE == 0) {
        // C^T layout: lane column = m, VGPR r walks n. One b128 store per tile.
        for (int ni = 0; ni < 2; ++ni)
            for (int mi = 0; mi < 2; ++mi) {
                int m  = m0 + m_base + mi * 16 + lrow;
                int n8 = n0 + n_base + ni * 16 + half * 8;   // 8-aligned
                v8f bvec = *(const v8f*)&bias[n8];
                pack8 p;
                for (int r = 0; r < 8; ++r)
                    p.h[r] = (bf16_t)(acc[ni][mi][r] + bvec[r]);
                *(v4u*)&out[(size_t)m * D_DIM + n8] = p.u4;
            }
    } else {
        // C layout: lane column = n (dim), VGPR r walks m -> l contiguously.
        for (int mi = 0; mi < 2; ++mi)
            for (int ni = 0; ni < 2; ++ni) {
                int n  = n0 + n_base + ni * 16 + lrow;
                int m8 = m0 + m_base + mi * 16 + half * 8;   // 8-aligned
                int b = m8 / L_DIM, l0 = m8 % L_DIM;
                float bv = bias[n];
                pack8 p;
                for (int r = 0; r < 8; ++r)
                    p.h[r] = (bf16_t)(acc[mi][ni][r] + bv);
                *(v4u*)&out[((size_t)b * D_DIM + n) * L_DIM + l0] = p.u4;
            }
    }
}

// ---------------------------------------------------------------------------
// Attention: one block (4 waves) per 16-query tile. LDS: Q tile 32KB +
// fp32 scores 16x2048 128KB + bf16 P 64KB + reduction scratch (~225KB of
// the WGP's 320KB).
// ---------------------------------------------------------------------------
__global__ __launch_bounds__(128) void attention_kernel(
    const bf16_t* __restrict__ qb, const bf16_t* __restrict__ kb,
    const bf16_t* __restrict__ vt, const float* __restrict__ mask,
    float* __restrict__ out)
{
    extern __shared__ char smem[];
    bf16_t* q_lds = (bf16_t*)smem;                          // 16*1024*2 = 32768
    float*  s_lds = (float*)(smem + 32768);                 // 16*2048*4 = 131072
    bf16_t* p_lds = (bf16_t*)(smem + 32768 + 131072);       // 16*2048*2 = 65536
    float*  red    = (float*)(smem + 32768 + 131072 + 65536); // 128 floats
    float*  rowmax = red + 128;                               // 16
    float*  rowsum = rowmax + 16;                             // 16

    const int tid  = threadIdx.x;
    const int lane = tid & 31;
    const int wave = tid >> 5;
    const int half = lane >> 4;
    const int lrow = lane & 15;

    const int b  = blockIdx.x >> 7;          // 128 query tiles per batch
    const int q0 = (blockIdx.x & 127) * 16;

    // load Q tile (16 x 1024 bf16 = 2048 x 16B) cooperatively, b128 copies
    {
        const v4u* src = (const v4u*)(qb + ((size_t)b * L_DIM + q0) * D_DIM);
        v4u* dst = (v4u*)q_lds;
        for (int i = tid; i < 16 * D_DIM / 8; i += 128) dst[i] = src[i];
    }
    __syncthreads();

    // ---- phase 1: S = (Q K^T) * scale + maskterm, full 2048 keys -----------
    for (int kt = wave; kt < L_DIM / 16; kt += 4) {
        int key0 = kt * 16;
        v8f acc = {};
        // B fragment: lane = key column, reads K[key][d..d+15] contiguous
        const bf16_t* kbase =
            kb + ((size_t)b * L_DIM + key0 + lrow) * D_DIM + half * 16;
        const bf16_t* qrow = q_lds + (size_t)lrow * D_DIM;
        for (int dk = 0; dk < D_DIM / 32; ++dk) {
            __builtin_prefetch(kbase + dk * 32 + 256, 0, 1);
            union { v16bf v; v8bf h[2]; } ua;
            ua.v = *(const v16bf*)&qrow[dk * 32 + half * 16];
            v16bf bf = *(const v16bf*)(kbase + dk * 32);
            acc = __builtin_amdgcn_wmma_f32_16x16x32_bf16(
                false, ua.v, false, bf, (short)0, acc, false, false);
        }
        int keycol = key0 + lrow;
        float mterm = (1.0f - mask[(size_t)b * L_DIM + keycol]) * -10000.0f;
        for (int r = 0; r < 8; ++r)
            s_lds[(r + half * 8) * L_DIM + keycol] = acc[r] * SCORE_SCALE + mterm;
    }
    __syncthreads();

    // ---- softmax: row max ---------------------------------------------------
    {
        int r = tid >> 3, c = tid & 7;
        float mx = -3.0e38f;
        for (int k = c * 256; k < c * 256 + 256; ++k)
            mx = fmaxf(mx, s_lds[r * L_DIM + k]);
        red[r * 8 + c] = mx;
    }
    __syncthreads();
    if (tid < 16) {
        float mx = red[tid * 8];
        for (int c = 1; c < 8; ++c) mx = fmaxf(mx, red[tid * 8 + c]);
        rowmax[tid] = mx;
    }
    __syncthreads();
    // ---- softmax: row sum of exp -------------------------------------------
    {
        int r = tid >> 3, c = tid & 7;
        float mx = rowmax[r], sum = 0.0f;
        for (int k = c * 256; k < c * 256 + 256; ++k)
            sum += __expf(s_lds[r * L_DIM + k] - mx);
        red[r * 8 + c] = sum;
    }
    __syncthreads();
    if (tid < 16) {
        float s = 0.0f;
        for (int c = 0; c < 8; ++c) s += red[tid * 8 + c];
        rowsum[tid] = 1.0f / s;
    }
    __syncthreads();
    // ---- P = softmax(S) in bf16, packed u32 writes --------------------------
    {
        unsigned int* p32 = (unsigned int*)p_lds;
        for (int i = tid; i < 16 * L_DIM / 2; i += 128) {
            int m = i >> 10, k2 = (i & 1023) * 2;
            float mx = rowmax[m], rs = rowsum[m];
            pack2 p;
            p.h[0] = (bf16_t)(__expf(s_lds[m * L_DIM + k2]     - mx) * rs);
            p.h[1] = (bf16_t)(__expf(s_lds[m * L_DIM + k2 + 1] - mx) * rs);
            p32[i] = p.u;
        }
    }
    __syncthreads();

    // ---- phase 2: O = P @ V. wave owns 256 output dims ---------------------
    for (int nt = 0; nt < 16; ++nt) {
        int d0 = wave * 256 + nt * 16;
        v8f acc = {};
        // B fragment: lane = dim column, reads Vt[d][key..key+15] contiguous
        const bf16_t* vbase =
            vt + ((size_t)b * D_DIM + d0 + lrow) * L_DIM + half * 16;
        const bf16_t* prow = p_lds + (size_t)lrow * L_DIM;
        for (int kblk = 0; kblk < L_DIM / 32; ++kblk) {
            __builtin_prefetch(vbase + kblk * 32 + 256, 0, 1);
            union { v16bf v; v8bf h[2]; } ua;
            ua.v = *(const v16bf*)&prow[kblk * 32 + half * 16];
            v16bf bf = *(const v16bf*)(vbase + kblk * 32);
            acc = __builtin_amdgcn_wmma_f32_16x16x32_bf16(
                false, ua.v, false, bf, (short)0, acc, false, false);
        }
        for (int r = 0; r < 8; ++r) {
            int m = q0 + r + half * 8;
            out[((size_t)b * L_DIM + m) * D_DIM + d0 + lrow] = acc[r];
        }
    }
}

// ---------------------------------------------------------------------------
extern "C" void kernel_launch(void* const* d_in, const int* in_sizes, int n_in,
                              void* d_out, int out_size, void* d_ws, size_t ws_size,
                              hipStream_t stream) {
    const float* q_in = (const float*)d_in[0];
    const float* k_in = (const float*)d_in[1];
    const float* v_in = (const float*)d_in[2];
    const float* mask = (const float*)d_in[3];
    const float* Wq   = (const float*)d_in[4];
    const float* bq   = (const float*)d_in[5];
    const float* Wk   = (const float*)d_in[6];
    const float* bk   = (const float*)d_in[7];
    const float* Wv   = (const float*)d_in[8];
    const float* bv   = (const float*)d_in[9];
    float* out = (float*)d_out;

    // workspace: Q bf16 (16MB) | K bf16 (16MB) | V^T bf16 (16MB)
    char* ws = (char*)d_ws;
    const size_t seg = (size_t)B_DIM * L_DIM * D_DIM * sizeof(bf16_t); // 16 MB
    bf16_t* qb = (bf16_t*)(ws);
    bf16_t* kb = (bf16_t*)(ws + seg);
    bf16_t* vt = (bf16_t*)(ws + 2 * seg);

    dim3 gemm_grid(D_DIM / 128, (B_DIM * L_DIM) / 64, 1); // (8, 128)
    qkv_gemm_kernel<0><<<gemm_grid, 256, 0, stream>>>(q_in, Wq, bq, qb);
    qkv_gemm_kernel<0><<<gemm_grid, 256, 0, stream>>>(k_in, Wk, bk, kb);
    qkv_gemm_kernel<1><<<gemm_grid, 256, 0, stream>>>(v_in, Wv, bv, vt);

    // dynamic LDS: 32768 + 131072 + 65536 + 160*4 = 230016 bytes (< 320KB WGP)
    const size_t lds_bytes = 32768 + 131072 + 65536 + 160 * sizeof(float);
    attention_kernel<<<dim3(B_DIM * L_DIM / 16), 128, lds_bytes, stream>>>(
        qb, kb, vt, mask, out);
}
